// GraphAttentionLayer_56307021250867
// MI455X (gfx1250) — matrix operand order repaired and checked by reading, not compiled, not run on hardware
//
#include <hip/hip_runtime.h>

#define NN   2048
#define FIN  512
#define FOUT 64
#define ALPHA 0.2f
#define BT    0.01f
#define NEG_BIG -9.0e15f

typedef __attribute__((ext_vector_type(16))) _Float16 v16h;
typedef __attribute__((ext_vector_type(8)))  float    v8f;

union V16H { v16h v; unsigned int u[8]; };

// leaky with slope in [0,1] == max(x, slope*x)  -> v_max_f32
__device__ __forceinline__ float leaky01(float x) { return fmaxf(x, BT * x); }

// ---------------- K0: f32 -> f16 conversions -------------------------------
__global__ void k_cvt_x(const float* __restrict__ x, _Float16* __restrict__ xh) {
    int i = blockIdx.x * blockDim.x + threadIdx.x;
    if (i < NN * FIN) xh[i] = (_Float16)x[i];
}

__global__ void k_cvt_w(const float* __restrict__ W, _Float16* __restrict__ WhT) {
    int i = blockIdx.x * blockDim.x + threadIdx.x;
    if (i < FIN * FOUT) {
        int k = i / FOUT, c = i % FOUT;
        WhT[c * FIN + k] = (_Float16)W[i];
    }
}

// ---------------- K1: h = leaky01(x @ W) via WMMA --------------------------
__global__ void k_h_gemm(const _Float16* __restrict__ xh, const _Float16* __restrict__ WhT,
                         float* __restrict__ h, _Float16* __restrict__ hT) {
    const int lane = threadIdx.x & 31;
    const int wave = threadIdx.x >> 5;
    const int l = lane & 15, half = lane >> 4;
    const int m0 = blockIdx.x * 16;
    const int n0 = wave * 16;

    v8f c = {};
    for (int k0 = 0; k0 < FIN; k0 += 32) {
        V16H a, b;
        const int m = m0 + l;
#pragma unroll
        for (int v = 0; v < 8; ++v) {          // A: 16x32 f16, ISA lane layout
            int kb = k0 + ((v < 4) ? 0 : 16) + 8 * half + 2 * (v & 3);
            a.u[v] = *(const unsigned int*)(xh + m * FIN + kb);
        }
        const int cc = n0 + l;
#pragma unroll
        for (int j = 0; j < 8; ++j) {          // B: 32x16 f16, K pairs contiguous in WhT
            int kb = k0 + ((j < 4) ? 0 : 16) + 8 * half + 2 * (j & 3);
            b.u[j] = *(const unsigned int*)(WhT + cc * FIN + kb);
        }
        c = __builtin_amdgcn_wmma_f32_16x16x32_f16(false, a.v, false, b.v,
                                                   (short)0, c, false, false);
    }
#pragma unroll
    for (int i = 0; i < 8; ++i) {              // C/D: M = i + 8*half, N = l
        int m = m0 + i + 8 * half;
        int n = n0 + l;
        float val = leaky01(c[i]);
        h[m * FOUT + n] = val;
        hT[n * NN + m] = (_Float16)val;
    }
}

// ---------------- K2: per-row score scalars --------------------------------
// q[i] = h_i@a1 + 0.505*(h_i@a3 + h_i@a4);  r[j] = h_j@a2 + 0.505*(h_j@a3 - h_j@a4)
__global__ void k_proj(const float* __restrict__ h, const float* __restrict__ a,
                       float* __restrict__ qr) {
    int i = blockIdx.x * blockDim.x + threadIdx.x;
    if (i >= NN) return;
    float s1 = 0.f, s2 = 0.f, s3 = 0.f, s4 = 0.f;
#pragma unroll 8
    for (int k = 0; k < FOUT; ++k) {
        float hv = h[i * FOUT + k];
        s1 += hv * a[k];
        s2 += hv * a[FOUT + k];
        s3 += hv * a[2 * FOUT + k];
        s4 += hv * a[3 * FOUT + k];
    }
    qr[i]      = s1 + 0.505f * (s3 + s4);
    qr[NN + i] = s2 + 0.505f * (s3 - s4);
}

// ---------------- K3: scores + mask + row softmax --------------------------
// one block (64 threads) per row i.  h chunks (64 rows) are double-buffered in
// LDS via async global->LDS b128 copies (ASYNCcnt); all hot-loop LDS reads are
// b128; nonlinear part is |.|-fma only (abs = free src modifier).
#define K3_T   64                 // threads / block
#define K3_R   64                 // rows / chunk
#define K3_Q   17                 // float4 per padded row (68 floats, 16B aligned)
#define K3_CH  (NN / K3_R)        // 32 chunks

__global__ void k_scores(const float* __restrict__ h, const float* __restrict__ qr,
                         const float* __restrict__ a, const int* __restrict__ adj,
                         _Float16* __restrict__ att) {
    __shared__ float4 hs4[2][K3_R * K3_Q];
    __shared__ float4 hi4[FOUT / 4];
    __shared__ float4 a3q[FOUT / 4];
    __shared__ float4 a4q[FOUT / 4];
    __shared__ float  red[K3_T];

    const int i = blockIdx.x;
    const int tid = threadIdx.x;

    if (tid < FOUT / 4) {
        hi4[tid] = *(const float4*)(h + i * FOUT + 4 * tid);
        a3q[tid] = *(const float4*)(a + 2 * FOUT + 4 * tid);
        a4q[tid] = *(const float4*)(a + 3 * FOUT + 4 * tid);
    }

    // LDS byte addresses of the two staging buffers (flat-addr low 32 bits)
    const unsigned lds_base0 = (unsigned)(unsigned long long)(&hs4[0][0]);
    const unsigned lds_base1 = (unsigned)(unsigned long long)(&hs4[1][0]);

    // issue async copy of chunk `cch` into buffer `buf` (16 b128 per thread)
    auto stage = [&](int cch, int buf) {
        unsigned base = buf ? lds_base1 : lds_base0;
#pragma unroll
        for (int ii = 0; ii < (K3_R * 16) / K3_T; ++ii) {
            int idx = tid + ii * K3_T;              // 0..1023
            int r = idx >> 4, q = idx & 15;         // row, quad
            unsigned laddr = base + (unsigned)(r * K3_Q + q) * 16u;
            const float* gaddr = h + (size_t)(cch * K3_R + r) * FOUT + 4 * q;
            asm volatile("global_load_async_to_lds_b128 %0, %1, off"
                         :: "v"(laddr), "v"(gaddr) : "memory");
        }
    };

    stage(0, 0);
    asm volatile("s_wait_asynccnt 0x0" ::: "memory");
    __syncthreads();

    const float qi = qr[i];
    float sc[K3_CH];

    for (int cch = 0; cch < K3_CH; ++cch) {
        if (cch + 1 < K3_CH) stage(cch + 1, (cch + 1) & 1);   // prefetch next chunk

        const float4* Hrow = &hs4[cch & 1][tid * K3_Q];
        float s = 0.f;
#pragma unroll
        for (int q = 0; q < FOUT / 4; ++q) {
            float4 H = Hrow[q];
            float4 I = hi4[q];
            float4 A3 = a3q[q];
            float4 A4 = a4q[q];
            s = fmaf(fabsf(I.x + H.x), A3.x, s);  s = fmaf(fabsf(I.x - H.x), A4.x, s);
            s = fmaf(fabsf(I.y + H.y), A3.y, s);  s = fmaf(fabsf(I.y - H.y), A4.y, s);
            s = fmaf(fabsf(I.z + H.z), A3.z, s);  s = fmaf(fabsf(I.z - H.z), A4.z, s);
            s = fmaf(fabsf(I.w + H.w), A3.w, s);  s = fmaf(fabsf(I.w - H.w), A4.w, s);
        }
        int j = cch * K3_R + tid;
        s = qi + qr[NN + j] + 0.495f * s;
        float e = fmaxf(s, ALPHA * s);                        // leaky(0.2)
        sc[cch] = (adj[i * NN + j] > 0) ? e : NEG_BIG;

        asm volatile("s_wait_asynccnt 0x0" ::: "memory");     // next chunk resident
        __syncthreads();                                      // visible to both waves
    }

    // row max
    float mx = sc[0];
#pragma unroll
    for (int t = 1; t < K3_CH; ++t) mx = fmaxf(mx, sc[t]);
    red[tid] = mx;
    __syncthreads();
    for (int off = K3_T / 2; off > 0; off >>= 1) {
        if (tid < off) red[tid] = fmaxf(red[tid], red[tid + off]);
        __syncthreads();
    }
    mx = red[0];
    __syncthreads();

    // exp + row sum
    float sum = 0.f;
#pragma unroll
    for (int t = 0; t < K3_CH; ++t) { sc[t] = __expf(sc[t] - mx); sum += sc[t]; }
    red[tid] = sum;
    __syncthreads();
    for (int off = K3_T / 2; off > 0; off >>= 1) {
        if (tid < off) red[tid] += red[tid + off];
        __syncthreads();
    }
    float inv = 1.f / red[0];

#pragma unroll
    for (int t = 0; t < K3_CH; ++t) {
        int j = t * K3_R + tid;
        att[(size_t)i * NN + j] = (_Float16)(sc[t] * inv);    // row-major == WMMA A layout
    }
}

// ---------------- K4: out = leaky01(attention @ h) via WMMA ----------------
__global__ void k_out_gemm(const _Float16* __restrict__ att, const _Float16* __restrict__ hT,
                           float* __restrict__ out) {
    const int lane = threadIdx.x & 31;
    const int wave = threadIdx.x >> 5;
    const int l = lane & 15, half = lane >> 4;
    const int m0 = blockIdx.x * 16;
    const int n0 = wave * 16;

    v8f c = {};
    for (int k0 = 0; k0 < NN; k0 += 32) {
        V16H a, b;
        const size_t m = m0 + l;
#pragma unroll
        for (int v = 0; v < 8; ++v) {
            int kb = k0 + ((v < 4) ? 0 : 16) + 8 * half + 2 * (v & 3);
            a.u[v] = *(const unsigned int*)(att + m * NN + kb);
        }
        const int cc = n0 + l;
#pragma unroll
        for (int j = 0; j < 8; ++j) {
            int kb = k0 + ((j < 4) ? 0 : 16) + 8 * half + 2 * (j & 3);
            b.u[j] = *(const unsigned int*)(hT + cc * NN + kb);
        }
        c = __builtin_amdgcn_wmma_f32_16x16x32_f16(false, a.v, false, b.v,
                                                   (short)0, c, false, false);
    }
#pragma unroll
    for (int i = 0; i < 8; ++i) {
        int m = m0 + i + 8 * half;
        int n = n0 + l;
        out[m * FOUT + n] = leaky01(c[i]);
    }
}

// ---------------------------------------------------------------------------
extern "C" void kernel_launch(void* const* d_in, const int* in_sizes, int n_in,
                              void* d_out, int out_size, void* d_ws, size_t ws_size,
                              hipStream_t stream) {
    (void)in_sizes; (void)n_in; (void)out_size; (void)ws_size;
    const float* x   = (const float*)d_in[0];
    const int*   adj = (const int*)d_in[1];
    const float* W   = (const float*)d_in[2];
    const float* a   = (const float*)d_in[3];
    float* out = (float*)d_out;

    char* ws = (char*)d_ws;
    size_t off = 0;
    auto alloc = [&](size_t bytes) {
        void* ptr = ws + off;
        off = (off + bytes + 255) & ~(size_t)255;
        return ptr;
    };
    _Float16* xh  = (_Float16*)alloc((size_t)NN * FIN * 2);     // 2 MB
    _Float16* WhT = (_Float16*)alloc((size_t)FIN * FOUT * 2);   // 64 KB
    float*    h   = (float*)   alloc((size_t)NN * FOUT * 4);    // 512 KB
    _Float16* hT  = (_Float16*)alloc((size_t)FOUT * NN * 2);    // 256 KB
    float*    qr  = (float*)   alloc((size_t)2 * NN * 4);       // 16 KB
    _Float16* att = (_Float16*)alloc((size_t)NN * NN * 2);      // 8 MB

    k_cvt_x   <<<(NN * FIN + 255) / 256, 256, 0, stream>>>(x, xh);
    k_cvt_w   <<<(FIN * FOUT + 255) / 256, 256, 0, stream>>>(W, WhT);
    k_h_gemm  <<<NN / 16, 128, 0, stream>>>(xh, WhT, h, hT);
    k_proj    <<<NN / 256, 256, 0, stream>>>(h, a, qr);
    k_scores  <<<NN, K3_T, 0, stream>>>(h, qr, a, adj, att);
    k_out_gemm<<<NN / 16, 128, 0, stream>>>(att, hT, out);
}
